// MultiHeadAttentionBlock_5506148074046
// MI455X (gfx1250) — matrix operand order repaired and checked by reading, not compiled, not run on hardware
//
#include <hip/hip_runtime.h>
#include <hip/hip_bf16.h>

typedef __attribute__((ext_vector_type(16))) _Float16 v16h;
typedef __attribute__((ext_vector_type(8)))  float    v8f;
typedef _Float16 half_t;

typedef unsigned int u32x4 __attribute__((ext_vector_type(4)));
typedef int          i32x4 __attribute__((ext_vector_type(4)));
typedef int          i32x8 __attribute__((ext_vector_type(8)));

#define WMMA_F32_F16(a, b, c) \
  __builtin_amdgcn_wmma_f32_16x16x32_f16(false, (a), false, (b), (short)0, (c), false, false)

#if __has_builtin(__builtin_amdgcn_tensor_load_to_lds) && \
    __has_builtin(__builtin_amdgcn_s_wait_tensorcnt)
#define HAVE_TDM 1
#else
#define HAVE_TDM 0
#endif

// ---------------- fragment loaders ------------------------------------------

// A fragment 16x32 f16 from row-major src (lda in halves, 16B-aligned rows).
__device__ inline v16h frag_a_16x32(const half_t* __restrict__ src, int lda) {
  const int lane = threadIdx.x & 31;
  const int row  = lane & 15;
  const int khi  = (lane >> 4) << 4;  // 0 or 16
  union { uint4 u[2]; v16h v; } f;
  f.u[0] = *(const uint4*)(src + row * lda + khi);
  f.u[1] = *(const uint4*)(src + row * lda + khi + 8);
  return f.v;
}

// B fragment 32x16 f16 from fragment-order storage: [lane][16 halves].
__device__ inline v16h frag_b_lin(const half_t* __restrict__ fragbase) {
  const int lane = threadIdx.x & 31;
  union { uint4 u[2]; v16h v; } f;
  f.u[0] = *(const uint4*)(fragbase + lane * 16);
  f.u[1] = *(const uint4*)(fragbase + lane * 16 + 8);
  return f.v;
}

// ---------------- TDM 1-D copy (global -> LDS) -------------------------------

#if HAVE_TDM
typedef __attribute__((address_space(3))) char lds_char_t;

__device__ inline unsigned lds_byte_off(void* p) {
  return (unsigned)(size_t)(lds_char_t*)p;
}

// Copy n8 * 8 bytes, contiguous, global -> LDS via the Tensor Data Mover.
__device__ inline void tdm_load_1d(unsigned lds_byte, const void* gsrc, unsigned n8) {
  unsigned long long ga = (unsigned long long)(size_t)gsrc;
  u32x4 g0;
  g0[0] = 1u;                                   // count=1, is_restore=0
  g0[1] = lds_byte;                             // lds_addr
  g0[2] = (unsigned)ga;                         // global_addr[31:0]
  g0[3] = (unsigned)((ga >> 32) & 0x01FFFFFFu) | 0x80000000u;  // addr[56:32], type=2
  i32x8 g1;
  g1[0] = 0x00030000;                           // data_size=3 (8B), wg mask=0
  g1[1] = (int)(n8 << 16);                      // tensor_dim0[15:0]
  g1[2] = 0;
  g1[3] = (int)(n8 << 16);                      // tile_dim0
  g1[4] = 0; g1[5] = 0; g1[6] = 0; g1[7] = 0;
  i32x4 z4 = {0, 0, 0, 0};
#if __clang_major__ >= 23
  i32x8 z8 = {0, 0, 0, 0, 0, 0, 0, 0};
  __builtin_amdgcn_tensor_load_to_lds(g0, g1, z4, z4, z8, 0);
#else
  __builtin_amdgcn_tensor_load_to_lds(g0, g1, z4, z4, 0);
#endif
}
#endif

// ---------------- converters -------------------------------------------------

__global__ void cvt_f32_to_f16_kernel(const float* __restrict__ in,
                                      half_t* __restrict__ out, int n) {
  int i = blockIdx.x * blockDim.x + threadIdx.x;
  if (i < n) out[i] = (half_t)in[i];
}

// x[b][c][p] f32 -> xf in B-fragment order per batch (frag = kt*64 + ntile)
__global__ void cvt_x_to_bfrag_kernel(const float* __restrict__ x,
                                      half_t* __restrict__ xf) {
  int i = blockIdx.x * blockDim.x + threadIdx.x;  // 16*128*1024 elements
  const int p = i & 1023;
  const int c = (i >> 10) & 127;
  const int b = i >> 17;
  const int frag = ((c >> 5) << 6) + (p >> 4);
  const int ln   = (p & 15) | (((c >> 4) & 1) << 4);
  const int pos  = c & 15;
  xf[(((size_t)b * 256 + frag) * 512) + ln * 16 + pos] = (half_t)x[i];
}

// ---------------- QKV projection GEMM (no LDS, no barriers) ------------------
// M=1536 (o), N=1024 (p), K=128 per batch. Wave tile 16x64 (4 accumulators).

__global__ __launch_bounds__(256) void qkv_gemm_kernel(
    const half_t* __restrict__ wq, const half_t* __restrict__ xf,
    const float* __restrict__ b_qkv, half_t* __restrict__ qbuf,
    half_t* __restrict__ kbuf, half_t* __restrict__ vbuf) {
  const int t    = threadIdx.x;
  const int lane = t & 31;
  const int wid  = t >> 5;
  const int wm   = wid & 3;   // 4 waves in M (block M = 64)
  const int wn   = wid >> 2;  // 2 waves in N (block N = 128)
  const int pBase = blockIdx.x * 128;
  const int mBase = blockIdx.y * 64;
  const int b     = blockIdx.z;

  v8f acc[4];
#pragma unroll
  for (int i = 0; i < 4; ++i) { v8f z = {}; acc[i] = z; }
#pragma unroll
  for (int kt = 0; kt < 4; ++kt) {
    v16h af = frag_a_16x32(wq + (size_t)(mBase + wm * 16) * 128 + kt * 32, 128);
    const half_t* bb =
        xf + ((size_t)b * 256 + kt * 64 + (pBase >> 4) + wn * 4) * 512;
#pragma unroll
    for (int nt = 0; nt < 4; ++nt)
      acc[nt] = WMMA_F32_F16(af, frag_b_lin(bb + nt * 512), acc[nt]);
  }
  const int hi = (lane >> 4) << 3;
  const int ln = lane & 15;
#pragma unroll
  for (int nt = 0; nt < 4; ++nt) {
    const int p = pBase + wn * 64 + nt * 16 + ln;
#pragma unroll
    for (int r = 0; r < 8; ++r) {
      const int o = mBase + wm * 16 + r + hi;
      const float val = acc[nt][r] + b_qkv[o];
      const int head  = o / 384;
      const int which = (o / 128) % 3;
      const int cc    = o & 127;
      const int bh    = b * 4 + head;
      if (which == 0) {            // Q: row-major [bh][p][cc] (WMMA A operand)
        qbuf[((size_t)bh * 1024 + p) * 128 + cc] = (half_t)val;
      } else if (which == 1) {     // K: frag-order, frag = (p>>4)*4 + cc>>5
        const int frag = ((p >> 4) << 2) + (cc >> 5);
        const int l2   = (p & 15) | (((cc >> 4) & 1) << 4);
        kbuf[((size_t)bh * 256 + frag) * 512 + l2 * 16 + (cc & 15)] = (half_t)val;
      } else {                     // V: frag-order, frag = (p>>5)*8 + cc>>4
        const int frag = ((p >> 5) << 3) + (cc >> 4);
        const int l2   = (cc & 15) | (((p >> 4) & 1) << 4);
        vbuf[((size_t)bh * 256 + frag) * 512 + l2 * 16 + (p & 15)] = (half_t)val;
      }
    }
  }
}

// ---------------- fused flash attention --------------------------------------
// grid (8 q-supertiles, 64 bh); 8 waves x 16 q rows. K/V chunks are contiguous
// 8KB fragment blocks -> double-buffered in LDS, prefetched by the TDM.

__global__ __launch_bounds__(256) void attn_kernel(
    const half_t* __restrict__ qbuf, const half_t* __restrict__ kbuf,
    const half_t* __restrict__ vbuf, half_t* __restrict__ abuf) {
  __shared__ __align__(16) half_t ktile[2][8 * 512];  // 2 x 8 KB
  __shared__ __align__(16) half_t vtile[2][8 * 512];  // 2 x 8 KB
  __shared__ __align__(16) half_t Ps[8 * 16 * 32];    // per-wave P bounce
  const int t     = threadIdx.x;
  const int lane  = t & 31;
  const int wid   = t >> 5;
  const int bh    = blockIdx.y;
  const int qBase = blockIdx.x * 128 + wid * 16;
  const int ln    = lane & 15;
  const int hi    = (lane >> 4) << 3;

  const half_t* kbase = kbuf + (size_t)bh * 256 * 512;
  const half_t* vbase = vbuf + (size_t)bh * 256 * 512;

  v16h qf[4];
#pragma unroll
  for (int kt = 0; kt < 4; ++kt)
    qf[kt] = frag_a_16x32(&qbuf[((size_t)bh * 1024 + qBase) * 128 + kt * 32], 128);

  v8f oacc[8];
#pragma unroll
  for (int i = 0; i < 8; ++i) { v8f z = {}; oacc[i] = z; }
  float mrow[8], lrow[8];
#pragma unroll
  for (int r = 0; r < 8; ++r) { mrow[r] = -1e30f; lrow[r] = 0.f; }

  const float scale = 0.08838834764831845f;  // 1/sqrt(128)

#if HAVE_TDM
  const unsigned klds0 = lds_byte_off(&ktile[0][0]);
  const unsigned vlds0 = lds_byte_off(&vtile[0][0]);
  if (wid == 0) {  // prime the pipeline: stage j=0
    tdm_load_1d(klds0, kbase, 1024);
    tdm_load_1d(vlds0, vbase, 1024);
  }
#endif

  for (int j = 0; j < 32; ++j) {
    half_t* kt_cur = ktile[j & 1];
    half_t* vt_cur = vtile[j & 1];
#if HAVE_TDM
    if (wid == 0) {
      if (j < 31) {  // prefetch j+1 into the other buffer, then wait for j
        const unsigned nb = ((j + 1) & 1) << 13;  // 8 KB stride
        tdm_load_1d(klds0 + nb, kbase + (size_t)(8 * (j + 1)) * 512, 1024);
        tdm_load_1d(vlds0 + nb, vbase + (size_t)(8 * (j + 1)) * 512, 1024);
        __builtin_amdgcn_s_wait_tensorcnt(2);  // in-order: j's pair complete
      } else {
        __builtin_amdgcn_s_wait_tensorcnt(0);
      }
    }
#else
    { const uint4* ks = (const uint4*)(kbase + (size_t)(8 * j) * 512);
      const uint4* vs = (const uint4*)(vbase + (size_t)(8 * j) * 512);
      uint4* kd = (uint4*)kt_cur; uint4* vd = (uint4*)vt_cur;
      kd[t] = ks[t]; kd[t + 256] = ks[t + 256];
      vd[t] = vs[t]; vd[t + 256] = vs[t + 256]; }
#endif
    __syncthreads();

    // S = Q K^T : fragments at kt_cur[(nt*4 + kt)*512]
    v8f s0 = {}; v8f s1 = {};
#pragma unroll
    for (int kt = 0; kt < 4; ++kt) {
      v16h bk0 = frag_b_lin(&kt_cur[kt * 512]);
      v16h bk1 = frag_b_lin(&kt_cur[(4 + kt) * 512]);
      s0 = WMMA_F32_F16(qf[kt], bk0, s0);
      s1 = WMMA_F32_F16(qf[kt], bk1, s1);
    }

    // online softmax (row reductions stay inside 16-lane halves)
    float pr0[8], pr1[8];
#pragma unroll
    for (int r = 0; r < 8; ++r) {
      float v0 = s0[r] * scale, v1 = s1[r] * scale;
      float mx = fmaxf(v0, v1);
      mx = fmaxf(mx, __shfl_xor(mx, 1, 32));
      mx = fmaxf(mx, __shfl_xor(mx, 2, 32));
      mx = fmaxf(mx, __shfl_xor(mx, 4, 32));
      mx = fmaxf(mx, __shfl_xor(mx, 8, 32));
      const float mnew  = fmaxf(mrow[r], mx);
      const float alpha = __expf(mrow[r] - mnew);
      const float e0 = __expf(v0 - mnew);
      const float e1 = __expf(v1 - mnew);
      float rs = e0 + e1;
      rs += __shfl_xor(rs, 1, 32);
      rs += __shfl_xor(rs, 2, 32);
      rs += __shfl_xor(rs, 4, 32);
      rs += __shfl_xor(rs, 8, 32);
      lrow[r] = lrow[r] * alpha + rs;
      mrow[r] = mnew;
      pr0[r] = e0; pr1[r] = e1;
#pragma unroll
      for (int c2 = 0; c2 < 8; ++c2) oacc[c2][r] *= alpha;
    }

    // C-layout -> A-layout via per-wave LDS bounce (reads vectorize to b128)
    half_t* myP = &Ps[wid * 16 * 32];
#pragma unroll
    for (int r = 0; r < 8; ++r) {
      const int m = r + hi;
      myP[m * 32 + ln]      = (half_t)pr0[r];
      myP[m * 32 + 16 + ln] = (half_t)pr1[r];
    }
    __syncthreads();
    v16h pa = frag_a_16x32(myP, 32);
#pragma unroll
    for (int c2 = 0; c2 < 8; ++c2) {
      v16h bv = frag_b_lin(&vt_cur[c2 * 512]);
      oacc[c2] = WMMA_F32_F16(pa, bv, oacc[c2]);
    }
    __syncthreads();  // separates this buffer's reads from its next overwrite
  }

  // finalize into abuf, B-fragment order for the output GEMM
  const int bb   = bh >> 2;
  const int head = bh & 3;
#pragma unroll
  for (int r = 0; r < 8; ++r) {
    const float inv = 1.0f / lrow[r];
    const int p = qBase + r + hi;
#pragma unroll
    for (int c2 = 0; c2 < 8; ++c2) {
      const int cc   = head * 128 + c2 * 16 + ln;   // 0..511
      const int frag = ((cc >> 5) << 6) + (p >> 4); // kt*64 + ntile
      const int l2   = (p & 15) | (((cc >> 4) & 1) << 4);
      abuf[((size_t)bb * 1024 + frag) * 512 + l2 * 16 + (cc & 15)] =
          (half_t)(oacc[c2][r] * inv);
    }
  }
}

// ---------------- output projection + bias + residual ------------------------
// M=128, K=512, N=1024 per batch. Wave tile 16x64 (4 accumulators).

__global__ __launch_bounds__(256) void out_gemm_kernel(
    const half_t* __restrict__ wo, const half_t* __restrict__ abuf,
    const float* __restrict__ b_out, const float* __restrict__ x,
    float* __restrict__ out) {
  const int t    = threadIdx.x;
  const int lane = t & 31;
  const int wid  = t >> 5;
  const int wm   = wid & 3;
  const int wn   = wid >> 2;
  const int pBase = blockIdx.x * 128;
  const int mBase = blockIdx.y * 64;
  const int b     = blockIdx.z;

  v8f acc[4];
#pragma unroll
  for (int i = 0; i < 4; ++i) { v8f z = {}; acc[i] = z; }
#pragma unroll 4
  for (int kt = 0; kt < 16; ++kt) {
    v16h af = frag_a_16x32(wo + (size_t)(mBase + wm * 16) * 512 + kt * 32, 512);
    const half_t* bb =
        abuf + ((size_t)b * 1024 + kt * 64 + (pBase >> 4) + wn * 4) * 512;
#pragma unroll
    for (int nt = 0; nt < 4; ++nt)
      acc[nt] = WMMA_F32_F16(af, frag_b_lin(bb + nt * 512), acc[nt]);
  }
  const int hi = (lane >> 4) << 3;
  const int ln = lane & 15;
#pragma unroll
  for (int nt = 0; nt < 4; ++nt) {
    const int p = pBase + wn * 64 + nt * 16 + ln;
#pragma unroll
    for (int r = 0; r < 8; ++r) {
      const int o = mBase + wm * 16 + r + hi;
      const size_t idx = ((size_t)b * 128 + o) * 1024 + p;
      out[idx] = acc[nt][r] + b_out[o] + x[idx];
    }
  }
}

// ---------------- launch -----------------------------------------------------

extern "C" void kernel_launch(void* const* d_in, const int* in_sizes, int n_in,
                              void* d_out, int out_size, void* d_ws, size_t ws_size,
                              hipStream_t stream) {
  (void)in_sizes; (void)n_in; (void)out_size; (void)ws_size;
  const float* x     = (const float*)d_in[0];  // 16x128x32x32
  const float* w_qkv = (const float*)d_in[1];  // 1536x128
  const float* b_qkv = (const float*)d_in[2];  // 1536
  const float* w_out = (const float*)d_in[3];  // 128x512
  const float* b_out = (const float*)d_in[4];  // 128
  float* out = (float*)d_out;

  char* ws = (char*)d_ws;
  half_t* xf  = (half_t*)(ws);                       // frag-order x,  4 MB
  half_t* wqf = (half_t*)(ws + (4u << 20));          // row-major,   384 KB
  half_t* wof = (half_t*)(ws + (4u << 20) + 393216); // row-major,   128 KB
  size_t off = (4u << 20) + 393216 + 131072;
  half_t* qbuf = (half_t*)(ws + off); off += (16u << 20);  // Q row-major
  half_t* kbuf = (half_t*)(ws + off); off += (16u << 20);  // K frag-order
  half_t* vbuf = (half_t*)(ws + off); off += (16u << 20);  // V frag-order
  half_t* abuf = (half_t*)(ws + off);                      // attn out frag-order

  cvt_x_to_bfrag_kernel<<<2097152 / 256, 256, 0, stream>>>(x, xf);
  cvt_f32_to_f16_kernel<<<196608 / 256, 256, 0, stream>>>(w_qkv, wqf, 196608);
  cvt_f32_to_f16_kernel<<<65536 / 256, 256, 0, stream>>>(w_out, wof, 65536);

  qkv_gemm_kernel<<<dim3(8, 24, 16), 256, 0, stream>>>(wqf, xf, b_qkv, qbuf, kbuf, vbuf);
  attn_kernel<<<dim3(8, 64), 256, 0, stream>>>(qbuf, kbuf, vbuf, abuf);
  out_gemm_kernel<<<dim3(8, 2, 16), 256, 0, stream>>>(wof, abuf, b_out, x, out);
}
